// ALIGNNEncoder_64003602645158
// MI455X (gfx1250) — matrix operand order repaired and checked by reading, not compiled, not run on hardware
//
#include <hip/hip_runtime.h>
#include <hip/hip_bf16.h>

// ---------------------------------------------------------------------------
// ALIGNN encoder forward for gfx1250 (MI455X).
//  - All dense contractions: v_wmma_f32_16x16x32_f16 (wave32 WMMA).
//  - Weight tiles staged LDS<-global via TENSOR_LOAD_TO_LDS (TDM) +
//    s_wait_tensorcnt (CDNA5 async-tensor path).
//  - Irregular ops (segment mean/sum, BN stats) are VALU/atomic kernels.
// ---------------------------------------------------------------------------

typedef __attribute__((ext_vector_type(16))) _Float16     v16h;
typedef __attribute__((ext_vector_type(8)))  float        v8f;
typedef __attribute__((ext_vector_type(4)))  unsigned int u32x4;
typedef __attribute__((ext_vector_type(8)))  int          i32x8;
typedef __attribute__((ext_vector_type(4)))  int          i32x4;

#define NW 4            // waves per GEMM block (128 threads); 32 rows per wave
#define STAT_BLOCKS 256

enum { SRC_DENSE = 0, SRC_CONCAT2 = 1, SRC_EDGE = 2 };

__device__ __forceinline__ float silu_f(float v) {
    return v / (1.f + __expf(-v));
}

// Issue a TDM copy of 64*KPAD f16 elements from global `src` to LDS `ldsAddr`.
// Descriptor per CDNA5 ISA: 1D tensor, data_size=2B,
// tile_dim0 = tensor_dim0 = 64*KPAD (<= 10240, fits 16-bit tile field).
template <int KPAD>
__device__ __forceinline__ void tdm_load_wt(const _Float16* src, unsigned int ldsAddr)
{
    constexpr unsigned int L = 64u * (unsigned)KPAD;   // elements
    unsigned long long ga = (unsigned long long)(size_t)src;
    u32x4 g0 = { 1u,                                   // count=1, user descriptor
                 ldsAddr,                              // lds_addr [63:32]
                 (unsigned int)ga,                     // global_addr [95:64]
                 (unsigned int)((ga >> 32) & 0x01FFFFFFu) | (2u << 30) }; // addr hi | type=2
    i32x8 g1 = { (int)0x00010000u,                     // wg_mask=0, data_size=1 (2B)
                 (int)((L & 0xFFFFu) << 16),           // tensor_dim0[15:0]
                 (int)((L >> 16) | (1u << 16)),        // tensor_dim0[31:16], tensor_dim1=1
                 (int)(L << 16),                       // tile_dim0 = L
                 1,                                    // tile_dim1=1, tile_dim2=0
                 (int)L,                               // tensor_dim0_stride[31:0]
                 0, 0 };
    i32x4 gz = { 0, 0, 0, 0 };
#if defined(__clang_major__) && (__clang_major__ >= 23)
    i32x8 gz8 = { 0, 0, 0, 0, 0, 0, 0, 0 };
    __builtin_amdgcn_tensor_load_to_lds(g0, g1, gz, gz, gz8, 0);
#else
    __builtin_amdgcn_tensor_load_to_lds(g0, g1, gz, gz, 0);
#endif
}

// Pre-convert W[K,64] (f32) -> padded, transposed f16 wtg[64][KPAD].
__global__ __launch_bounds__(256) void prep_wt(
    const float* __restrict__ W, _Float16* __restrict__ wtg, int K, int KPAD_rt)
{
    int idx = blockIdx.x * blockDim.x + threadIdx.x;
    if (idx < 64 * KPAD_rt) {
        int col = idx / KPAD_rt, k = idx % KPAD_rt;
        wtg[idx] = (_Float16)((k < K) ? W[(long)k * 64 + col] : 0.f);
    }
}

// out[R,64] = op(A) @ W[K,64] + bias  (optionally SiLU on output).
// Per wave: M=32 (two 16-row tiles) x N=64, K in chunks of 32.
template <int KPAD, int MODE, bool PRE_BNSILU, bool POST_SILU>
__global__ __launch_bounds__(128) void wmma_gemm64(
    const float* __restrict__ A,        // dense:[R,K] / concat2:X[R,64] / edge: node feats [n,64]
    const float* __restrict__ A2,       // concat2: Y[R,64]
    const float* __restrict__ EA,       // edge: edge_attr [R,1]
    const int*   __restrict__ srcIdx,   // edge: source node per row
    const int*   __restrict__ dstIdx,   // edge: dest node per row
    const _Float16* __restrict__ wtg,   // prepped W^T f16 [64][KPAD]
    const float* __restrict__ bias,     // [64]
    const float* __restrict__ affine,   // [128] = {scale[64], shift[64]} (PRE_BNSILU)
    float* __restrict__ out, int R, int K)
{
    __shared__ _Float16 Wt[64][KPAD];          // W transposed: Wt[col][k]
    __shared__ _Float16 At[NW][32][KPAD + 8];  // per-wave 32-row A tile (f16)

    const int tid  = threadIdx.x;
    const int lane = tid & 31;
    const int wave = tid >> 5;
    const int rowTile = blockIdx.x * (NW * 32) + wave * 32;

    // Kick off TDM weight-tile DMA (wave 0 only; EXEC ignored by TDM).
    if (wave == 0)
        tdm_load_wt<KPAD>(wtg, (unsigned int)(size_t)&Wt[0][0]);

    // Stage this wave's 32-row A tile (gather/concat + optional BN+SiLU),
    // overlapped with the TDM transfer.
    for (int idx = lane; idx < 32 * KPAD; idx += 32) {
        int r = idx / KPAD, k = idx % KPAD;
        int m = rowTile + r;
        float v = 0.f;
        if (m < R && k < K) {
            if (MODE == SRC_DENSE) {
                v = A[(long)m * K + k];
            } else if (MODE == SRC_CONCAT2) {
                v = (k < 64) ? A[(long)m * 64 + k] : A2[(long)m * 64 + (k - 64)];
            } else { // SRC_EDGE: concat(h[dst], h[src], ea)
                if (k < 64)        v = A[(long)dstIdx[m] * 64 + k];
                else if (k < 128)  v = A[(long)srcIdx[m] * 64 + (k - 64)];
                else if (k == 128) v = EA[m];
            }
            if (PRE_BNSILU) {
                v = v * affine[k] + affine[64 + k];
                v = silu_f(v);
            }
        }
        At[wave][r][k] = (_Float16)v;
    }
    if (wave == 0)
        __builtin_amdgcn_s_wait_tensorcnt(0);
    __syncthreads();

    // Per-lane K index maps (ISA 7.12.2, 16-bit A 16x32 and B 32x16 layouts).
    const int hi = lane >> 4;               // 0 for lanes 0-15, 1 for 16-31
    int kA[16], kB[16];
#pragma unroll
    for (int i = 0; i < 16; ++i) {
        kA[i] = i + 8 * hi + ((i >= 8) ? 8 : 0);
        kB[i] = i + 8 * hi + ((i >= 8) ? 8 : 0);
    }

    v8f zero = {};
    v8f acc[2][4];
#pragma unroll
    for (int t = 0; t < 2; ++t)
#pragma unroll
        for (int n = 0; n < 4; ++n) acc[t][n] = zero;

    const int arow = lane & 15;
#pragma unroll
    for (int kb = 0; kb < KPAD; kb += 32) {
        v16h afrag0, afrag1;
#pragma unroll
        for (int i = 0; i < 16; ++i) {
            afrag0[i] = At[wave][arow][kb + kA[i]];
            afrag1[i] = At[wave][16 + arow][kb + kA[i]];
        }
#pragma unroll
        for (int n = 0; n < 4; ++n) {
            v16h bfrag;
            const int col = n * 16 + arow;
#pragma unroll
            for (int i = 0; i < 16; ++i) bfrag[i] = Wt[col][kb + kB[i]];
            acc[0][n] = __builtin_amdgcn_wmma_f32_16x16x32_f16(
                false, afrag0, false, bfrag, (short)0, acc[0][n], false, false);
            acc[1][n] = __builtin_amdgcn_wmma_f32_16x16x32_f16(
                false, afrag1, false, bfrag, (short)0, acc[1][n], false, false);
        }
    }

    // C/D layout: VGPR r -> M = r + 8*hi, N = lane%16 (+16*n tile).
#pragma unroll
    for (int t = 0; t < 2; ++t) {
        const int mbase = rowTile + t * 16 + (hi << 3);
#pragma unroll
        for (int n = 0; n < 4; ++n) {
            const int c = n * 16 + arow;
            const float bv = bias[c];
#pragma unroll
            for (int r = 0; r < 8; ++r) {
                int m = mbase + r;
                if (m < R) {
                    float v = acc[t][n][r] + bv;
                    if (POST_SILU) v = silu_f(v);
                    out[(long)m * 64 + c] = v;
                }
            }
        }
    }
}

// ---------------- BatchNorm statistics (deterministic two-pass) -------------

__global__ __launch_bounds__(256) void bn_stats_partial(
    const float* __restrict__ X, long R, float* __restrict__ part)
{
    __shared__ float sh[512];
    const int c = threadIdx.x & 63;
    const int g = threadIdx.x >> 6;     // 4 row groups
    float s = 0.f, q = 0.f;
    for (long r = (long)blockIdx.x * 4 + g; r < R; r += (long)STAT_BLOCKS * 4) {
        float v = X[r * 64 + c];
        s += v; q += v * v;
    }
    sh[threadIdx.x] = s;
    sh[256 + threadIdx.x] = q;
    __syncthreads();
    if (g == 0) {
        for (int gg = 1; gg < 4; ++gg) {
            s += sh[gg * 64 + c];
            q += sh[256 + gg * 64 + c];
        }
        part[blockIdx.x * 128 + c]      = s;
        part[blockIdx.x * 128 + 64 + c] = q;
    }
}

__global__ __launch_bounds__(64) void bn_finalize(
    const float* __restrict__ part, long R,
    const float* __restrict__ gamma, const float* __restrict__ beta,
    float* __restrict__ affine)
{
    const int c = threadIdx.x;
    float s = 0.f, q = 0.f;
    for (int i = 0; i < STAT_BLOCKS; ++i) {
        s += part[i * 128 + c];
        q += part[i * 128 + 64 + c];
    }
    const float fR = (float)R;
    const float mean = s / fR;
    const float var  = q / fR - mean * mean;
    const float sc   = gamma[c] * rsqrtf(var + 1e-5f);
    affine[c]      = sc;
    affine[64 + c] = beta[c] - mean * sc;
}

__global__ __launch_bounds__(256) void bn_silu_apply(
    const float* __restrict__ X, const float* __restrict__ affine,
    float* __restrict__ Y, long n)
{
    long i = (long)blockIdx.x * blockDim.x + threadIdx.x;
    if (i < n) {
        int c = (int)(i & 63);
        float v = X[i] * affine[c] + affine[64 + c];
        Y[i] = silu_f(v);
    }
}

// ---------------- segment ops ----------------------------------------------

__global__ __launch_bounds__(256) void zero_f32(float* __restrict__ p, long n)
{
    for (long i = (long)blockIdx.x * blockDim.x + threadIdx.x; i < n;
         i += (long)gridDim.x * blockDim.x)
        p[i] = 0.f;
}

// agg[d,c] += M[e,c] where d = idx2 ? idx2[idx[e]] : idx[e]; optional counts.
__global__ __launch_bounds__(256) void scatter_add64(
    const float* __restrict__ M, const int* __restrict__ idx,
    const int* __restrict__ idx2,
    float* __restrict__ agg, float* __restrict__ cnt, long R, int withCount)
{
    long i = (long)blockIdx.x * blockDim.x + threadIdx.x;
    if (i < R * 64) {
        long e = i >> 6;
        int  c = (int)(i & 63);
        int  d = idx[e];
        if (idx2) d = idx2[d];
        atomicAdd(&agg[(long)d * 64 + c], M[i]);
        if (withCount && c == 0) atomicAdd(&cnt[d], 1.0f);
    }
}

__global__ __launch_bounds__(256) void seg_div64(
    float* __restrict__ agg, const float* __restrict__ cnt, long n)
{
    long i = (long)blockIdx.x * blockDim.x + threadIdx.x;
    if (i < n) agg[i] /= fmaxf(cnt[i >> 6], 1.0f);
}

// ---------------------------------------------------------------------------

extern "C" void kernel_launch(void* const* d_in, const int* in_sizes, int n_in,
                              void* d_out, int out_size, void* d_ws, size_t ws_size,
                              hipStream_t stream)
{
    (void)in_sizes; (void)n_in; (void)out_size; (void)ws_size;
    constexpr int N = 20000, E = 320000, LE = 640000, Bg = 200;

    const float* x     = (const float*)d_in[0];
    const float* eatt  = (const float*)d_in[1];
    const float* lx    = (const float*)d_in[2];
    const float* leatt = (const float*)d_in[3];
    const float* Wa    = (const float*)d_in[4];
    const float* ba    = (const float*)d_in[5];
    const float* Wl    = (const float*)d_in[6];
    const float* bl    = (const float*)d_in[7];
    const float* aW1   = (const float*)d_in[8];
    const float* ab1   = (const float*)d_in[9];
    const float* ag1   = (const float*)d_in[10];
    const float* abe1  = (const float*)d_in[11];
    const float* aW2   = (const float*)d_in[12];
    const float* ab2   = (const float*)d_in[13];
    const float* aWu   = (const float*)d_in[14];
    const float* abu   = (const float*)d_in[15];
    const float* agu   = (const float*)d_in[16];
    const float* abeu  = (const float*)d_in[17];
    const float* lW1   = (const float*)d_in[18];
    const float* lb1   = (const float*)d_in[19];
    const float* lg1   = (const float*)d_in[20];
    const float* lbe1  = (const float*)d_in[21];
    const float* lW2   = (const float*)d_in[22];
    const float* lb2   = (const float*)d_in[23];
    const float* lWu   = (const float*)d_in[24];
    const float* lbu   = (const float*)d_in[25];
    const float* lgu   = (const float*)d_in[26];
    const float* lbeu  = (const float*)d_in[27];
    const float* bWf   = (const float*)d_in[28];
    const float* bbf   = (const float*)d_in[29];
    const float* bgf   = (const float*)d_in[30];
    const float* bbef  = (const float*)d_in[31];
    const float* Wo    = (const float*)d_in[32];
    const float* bo    = (const float*)d_in[33];
    const int*   eidx  = (const int*)d_in[34];
    const int*   leidx = (const int*)d_in[35];
    const int*   batch = (const int*)d_in[36];

    const int* src  = eidx;          // atom edges: source
    const int* dst  = eidx + E;      // atom edges: dest
    const int* lsrc = leidx;         // line edges: source
    const int* ldst = leidx + LE;    // line edges: dest

    // ---- workspace carve-up (floats) ----
    float* ws = (float*)d_ws;
    size_t off = 0;
    auto alloc = [&](size_t n) { float* p = ws + off; off += n; return p; };
    float* h    = alloc((size_t)N * 64);
    float* auxN = alloc((size_t)N * 64);   // agg_atom / abf
    float* hU   = alloc((size_t)N * 64);   // atom pre-BN update
    float* lh   = alloc((size_t)E * 64);
    float* auxE = alloc((size_t)E * 64);   // agg_line
    float* ebuf = alloc((size_t)LE * 64);  // edge messages (reused in place) + line update
    float* cntN = alloc((size_t)N);
    float* cntE = alloc((size_t)E);
    float* cntB = alloc((size_t)Bg);
    float* cntB2= alloc((size_t)Bg);
    float* embA = alloc((size_t)Bg * 64);
    float* embL = alloc((size_t)Bg * 64);
    float* part = alloc((size_t)STAT_BLOCKS * 128);
    float* aff  = alloc(128);
    _Float16* wtg = (_Float16*)alloc(64 * 160 / 2 + 64);  // f16 W^T staging (<=64*160 halves)

    auto cdiv = [](long a, long b) { return (int)((a + b - 1) / b); };
    auto prep = [&](const float* W, int K, int KPAD) {
        prep_wt<<<(64 * KPAD) / 256, 256, 0, stream>>>(W, wtg, K, KPAD);
    };

    // One MessagePassing conv layer (mean aggregation) on `nodes` [nN,64].
    auto conv = [&](float* nodes, int nN, long nE,
                    const int* s, const int* dd, const float* ea,
                    const float* W1, const float* b1, const float* g1, const float* be1,
                    const float* W2, const float* b2,
                    const float* Wu, const float* bu, const float* gu, const float* beu,
                    float* mbuf, float* agg, float* cnt, float* U) {
        const int gE = cdiv(nE, NW * 32);
        // m1 = concat(x[dst], x[src], ea) @ W1 + b1
        prep(W1, 129, 160);
        wmma_gemm64<160, SRC_EDGE, false, false><<<gE, 128, 0, stream>>>(
            nodes, nullptr, ea, s, dd, wtg, b1, nullptr, mbuf, (int)nE, 129);
        bn_stats_partial<<<STAT_BLOCKS, 256, 0, stream>>>(mbuf, nE, part);
        bn_finalize<<<1, 64, 0, stream>>>(part, nE, g1, be1, aff);
        // m2 = silu(bn(m1)) @ W2 + b2   (in place: block stages tile before write)
        prep(W2, 64, 64);
        wmma_gemm64<64, SRC_DENSE, true, false><<<gE, 128, 0, stream>>>(
            mbuf, nullptr, nullptr, nullptr, nullptr, wtg, b2, aff, mbuf, (int)nE, 64);
        // agg = segment_mean(m2, dst, nN)
        zero_f32<<<cdiv((long)nN * 64, 256), 256, 0, stream>>>(agg, (long)nN * 64);
        zero_f32<<<cdiv((long)nN, 256), 256, 0, stream>>>(cnt, (long)nN);
        scatter_add64<<<cdiv(nE * 64, 256), 256, 0, stream>>>(
            mbuf, dd, nullptr, agg, cnt, nE, 1);
        seg_div64<<<cdiv((long)nN * 64, 256), 256, 0, stream>>>(agg, cnt, (long)nN * 64);
        // nodes = silu(bn(concat(nodes, agg) @ Wu + bu))
        const int gN = cdiv(nN, NW * 32);
        prep(Wu, 128, 128);
        wmma_gemm64<128, SRC_CONCAT2, false, false><<<gN, 128, 0, stream>>>(
            nodes, agg, nullptr, nullptr, nullptr, wtg, bu, nullptr, U, nN, 128);
        bn_stats_partial<<<STAT_BLOCKS, 256, 0, stream>>>(U, (long)nN, part);
        bn_finalize<<<1, 64, 0, stream>>>(part, (long)nN, gu, beu, aff);
        bn_silu_apply<<<cdiv((long)nN * 64, 256), 256, 0, stream>>>(
            U, aff, nodes, (long)nN * 64);
    };

    // ---- initial embeddings ----
    prep(Wa, 92, 96);
    wmma_gemm64<96, SRC_DENSE, false, false><<<cdiv(N, NW * 32), 128, 0, stream>>>(
        x, nullptr, nullptr, nullptr, nullptr, wtg, ba, nullptr, h, N, 92);
    prep(Wl, 3, 32);
    wmma_gemm64<32, SRC_DENSE, false, false><<<cdiv(E, NW * 32), 128, 0, stream>>>(
        lx, nullptr, nullptr, nullptr, nullptr, wtg, bl, nullptr, lh, E, 3);

    const int gN = cdiv(N, NW * 32);

    // ---- joint layers (atom conv + line conv + bond-to-atom fusion) ----
    for (int i = 0; i < 3; ++i) {
        conv(h, N, E, src, dst, eatt,
             aW1 + (size_t)i * 129 * 64, ab1 + i * 64, ag1 + i * 64, abe1 + i * 64,
             aW2 + (size_t)i * 64 * 64,  ab2 + i * 64,
             aWu + (size_t)i * 128 * 64, abu + i * 64, agu + i * 64, abeu + i * 64,
             ebuf, auxN, cntN, hU);
        conv(lh, E, LE, lsrc, ldst, leatt,
             lW1 + (size_t)i * 129 * 64, lb1 + i * 64, lg1 + i * 64, lbe1 + i * 64,
             lW2 + (size_t)i * 64 * 64,  lb2 + i * 64,
             lWu + (size_t)i * 128 * 64, lbu + i * 64, lgu + i * 64, lbeu + i * 64,
             ebuf, auxE, cntE, ebuf /* line update reuses message buffer */);
        // abf = segment_sum(lh, src, N); h = silu(bn(concat(h, abf) @ bW + bb))
        zero_f32<<<cdiv((long)N * 64, 256), 256, 0, stream>>>(auxN, (long)N * 64);
        scatter_add64<<<cdiv((long)E * 64, 256), 256, 0, stream>>>(
            lh, src, nullptr, auxN, nullptr, (long)E, 0);
        prep(bWf + (size_t)i * 128 * 64, 128, 128);
        wmma_gemm64<128, SRC_CONCAT2, false, false><<<gN, 128, 0, stream>>>(
            h, auxN, nullptr, nullptr, nullptr,
            wtg, bbf + i * 64, nullptr, hU, N, 128);
        bn_stats_partial<<<STAT_BLOCKS, 256, 0, stream>>>(hU, (long)N, part);
        bn_finalize<<<1, 64, 0, stream>>>(part, (long)N, bgf + i * 64, bbef + i * 64, aff);
        bn_silu_apply<<<cdiv((long)N * 64, 256), 256, 0, stream>>>(
            hU, aff, h, (long)N * 64);
    }

    // ---- remaining atom-only layers ----
    for (int i = 3; i < 5; ++i) {
        conv(h, N, E, src, dst, eatt,
             aW1 + (size_t)i * 129 * 64, ab1 + i * 64, ag1 + i * 64, abe1 + i * 64,
             aW2 + (size_t)i * 64 * 64,  ab2 + i * 64,
             aWu + (size_t)i * 128 * 64, abu + i * 64, agu + i * 64, abeu + i * 64,
             ebuf, auxN, cntN, hU);
    }

    // ---- readout ----
    zero_f32<<<cdiv((long)Bg * 64, 256), 256, 0, stream>>>(embA, (long)Bg * 64);
    zero_f32<<<cdiv((long)Bg * 64, 256), 256, 0, stream>>>(embL, (long)Bg * 64);
    zero_f32<<<1, 256, 0, stream>>>(cntB, (long)Bg);
    zero_f32<<<1, 256, 0, stream>>>(cntB2, (long)Bg);
    scatter_add64<<<cdiv((long)N * 64, 256), 256, 0, stream>>>(
        h, batch, nullptr, embA, cntB, (long)N, 1);
    seg_div64<<<cdiv((long)Bg * 64, 256), 256, 0, stream>>>(embA, cntB, (long)Bg * 64);
    scatter_add64<<<cdiv((long)E * 64, 256), 256, 0, stream>>>(
        lh, src, batch, embL, cntB2, (long)E, 1);   // idx = batch[src[e]]
    seg_div64<<<cdiv((long)Bg * 64, 256), 256, 0, stream>>>(embL, cntB2, (long)Bg * 64);

    prep(Wo, 128, 128);
    wmma_gemm64<128, SRC_CONCAT2, false, true><<<cdiv(Bg, NW * 32), 128, 0, stream>>>(
        embA, embL, nullptr, nullptr, nullptr, wtg, bo, nullptr,
        (float*)d_out, Bg, 128);
}
